// Is_Stable_GNN_84894323572880
// MI455X (gfx1250) — matrix operand order repaired and checked by reading, not compiled, not run on hardware
//
#include <hip/hip_runtime.h>
#include <hip/hip_bf16.h>
#include <math.h>

typedef __attribute__((ext_vector_type(16))) _Float16 v16h;
typedef __attribute__((ext_vector_type(8)))  float    v8f;

#define Fdim  64
#define DEdim 32
#define Zdim  160      // 2F + DE
#define Gnum  512
#define EPSV  1e-5f

__device__ __forceinline__ void atomAddF(float* p, float v) {
    __hip_atomic_fetch_add(p, v, __ATOMIC_RELAXED, __HIP_MEMORY_SCOPE_AGENT);
}

__device__ __forceinline__ float softplusf(float v) {
    return fmaxf(v, 0.0f) + log1pf(__expf(-fabsf(v)));
}

// ---------------------------------------------------------------------------
// Kernel 0: zero workspace accumulators
// ---------------------------------------------------------------------------
__global__ void zero_f32(float* __restrict__ p, long n) {
    long i = (long)blockIdx.x * blockDim.x + threadIdx.x;
    long stride = (long)gridDim.x * blockDim.x;
    for (; i < n; i += stride) p[i] = 0.0f;
}

// ---------------------------------------------------------------------------
// Kernel 1: CGConv edge messages via WMMA f16 (f32 accumulate) + atomic scatter
//   One wave32 per 16-edge tile. A = z tile (gathered straight into a contiguous-K
//   A-fragment layout), B = W^T staged in LDS, 40 v_wmma per tile.
// ---------------------------------------------------------------------------
__global__ __launch_bounds__(256)
void edge_msg_wmma(const float* __restrict__ x,
                   const int* __restrict__ srcIdx,
                   const int* __restrict__ dstIdx,
                   const float* __restrict__ eattr,
                   const float* __restrict__ Wf, const float* __restrict__ bf,
                   const float* __restrict__ Ws, const float* __restrict__ bs,
                   float* __restrict__ agg, int E)
{
    // W^T in LDS: [col 0..63][k 0..159] halves -> B fragments are contiguous.
    __shared__ __attribute__((aligned(32))) _Float16 sWf[Fdim * Zdim];
    __shared__ __attribute__((aligned(32))) _Float16 sWs[Fdim * Zdim];

    const int tid = threadIdx.x;
    for (int i = tid; i < Zdim * Fdim; i += 256) {          // coalesced read, transposed store
        int k = i >> 6, c = i & 63;
        sWf[c * Zdim + k] = (_Float16)Wf[i];
        sWs[c * Zdim + k] = (_Float16)Ws[i];
    }
    __syncthreads();

    const int wave = tid >> 5;
    const int lane = tid & 31;
    const int e0 = (blockIdx.x * 8 + wave) * 16;
    if (e0 + 16 > E) return;                                // wave-uniform: EXEC stays all-1s

    const int mrow = lane & 15;     // A row owned by this lane
    const int kg   = lane >> 4;     // K-group (ISA A layout split)

    const long di = dstIdx[e0 + mrow];
    const long si = srcIdx[e0 + mrow];
    const float* xd = x + di * Fdim;                        // z[k]     k <  64 : x[dst]
    const float* xs = x + si * Fdim;                        // z[k] 64<=k < 128 : x[src]
    const float* ep = eattr + (long)(e0 + mrow) * DEdim;    // z[k] 128<=k<160 : edge_attr

    // Gather 5 A fragments. Lane<16 needs K {base..base+15};
    // lane>=16 needs K {base+16..base+31}. All 16-runs sit inside one region.
    v16h a[5];
#pragma unroll
    for (int kc = 0; kc < 5; ++kc) {
        const float* base; int off;
        int klo = kc * 32 + kg * 16;
        if (kc < 2)      { base = xd; off = klo; }
        else if (kc < 4) { base = xs; off = klo - 64; }
        else             { base = ep; off = klo - 128; }
        float4 l0 = *(const float4*)(base + off);
        float4 l1 = *(const float4*)(base + off + 4);
        float4 h0 = *(const float4*)(base + off + 8);
        float4 h1 = *(const float4*)(base + off + 12);
        v16h av;
        av[0]=(_Float16)l0.x; av[1]=(_Float16)l0.y; av[2]=(_Float16)l0.z; av[3]=(_Float16)l0.w;
        av[4]=(_Float16)l1.x; av[5]=(_Float16)l1.y; av[6]=(_Float16)l1.z; av[7]=(_Float16)l1.w;
        av[8]=(_Float16)h0.x; av[9]=(_Float16)h0.y; av[10]=(_Float16)h0.z; av[11]=(_Float16)h0.w;
        av[12]=(_Float16)h1.x; av[13]=(_Float16)h1.y; av[14]=(_Float16)h1.z; av[15]=(_Float16)h1.w;
        a[kc] = av;
    }

    // dst node ids for the 8 C rows this lane owns (C layout: M = kg*8 + r)
    int drow[8];
#pragma unroll
    for (int r = 0; r < 8; ++r) drow[r] = dstIdx[e0 + kg * 8 + r];

    const int kb0 = kg * 16;        // B fragment K offset for this lane half

#pragma unroll
    for (int nt = 0; nt < 4; ++nt) {
        const int col = nt * 16 + mrow;                     // output feature (= B column = C column)
        const float bfv = bf[col], bsv = bs[col];
        v8f accf = { bfv, bfv, bfv, bfv, bfv, bfv, bfv, bfv };
        v8f accs = { bsv, bsv, bsv, bsv, bsv, bsv, bsv, bsv };
#pragma unroll
        for (int kc = 0; kc < 5; ++kc) {
            const v16h bF = *(const v16h*)&sWf[col * Zdim + kc * 32 + kb0];
            accf = __builtin_amdgcn_wmma_f32_16x16x32_f16(false, a[kc], false, bF,
                                                          (short)0, accf, false, false);
            const v16h bS = *(const v16h*)&sWs[col * Zdim + kc * 32 + kb0];
            accs = __builtin_amdgcn_wmma_f32_16x16x32_f16(false, a[kc], false, bS,
                                                          (short)0, accs, false, false);
        }
        // message = sigmoid(lin_f) * softplus(lin_s); scatter-add into agg[dst]
#pragma unroll
        for (int r = 0; r < 8; ++r) {
            float fv = accf[r], sv = accs[r];
            float gsig = 1.0f / (1.0f + __expf(-fv));
            float msg  = gsig * softplusf(sv);
            atomAddF(&agg[(long)drow[r] * Fdim + col], msg);
        }
    }
}

// ---------------------------------------------------------------------------
// Kernel 2: per-feature-column sum / sumsq over agg (BatchNorm batch stats)
// ---------------------------------------------------------------------------
__global__ __launch_bounds__(256)
void col_stats(const float* __restrict__ agg, float* __restrict__ colsum,
               float* __restrict__ colsq, int N)
{
    __shared__ float ls[256], lq[256];
    int f  = threadIdx.x & 63;
    int rg = threadIdx.x >> 6;      // 4 row-groups per block
    float s = 0.0f, q = 0.0f;
    for (long i = (long)blockIdx.x * 4 + rg; i < N; i += (long)gridDim.x * 4) {
        float v = agg[i * 64 + f];
        s += v; q += v * v;
    }
    ls[threadIdx.x] = s; lq[threadIdx.x] = q;
    __syncthreads();
    if (rg == 0) {
        for (int j = 1; j < 4; ++j) { s += ls[f + 64 * j]; q += lq[f + 64 * j]; }
        atomAddF(&colsum[f], s);
        atomAddF(&colsq[f], q);
    }
}

// ---------------------------------------------------------------------------
// Kernel 3: BN(train stats) + residual; store h; accumulate whole-tensor
//           sum/sumsq for the PyG graph-mode LayerNorm
// ---------------------------------------------------------------------------
__global__ __launch_bounds__(256)
void bn_residual(const float* __restrict__ agg, const float* __restrict__ x,
                 const float* __restrict__ colsum, const float* __restrict__ colsq,
                 const float* __restrict__ bn_w, const float* __restrict__ bn_b,
                 float* __restrict__ h, float* __restrict__ lnS,
                 long total, float invN)
{
    __shared__ float ls[256], lq[256];
    float s = 0.0f, q = 0.0f;
    long stride = (long)gridDim.x * blockDim.x;
    for (long i = (long)blockIdx.x * blockDim.x + threadIdx.x; i < total; i += stride) {
        int f = (int)(i & 63);
        float mu  = colsum[f] * invN;
        float var = colsq[f] * invN - mu * mu;
        float hv = (agg[i] - mu) / sqrtf(var + EPSV) * bn_w[f] + bn_b[f] + x[i];
        h[i] = hv;
        s += hv; q += hv * hv;
    }
    ls[threadIdx.x] = s; lq[threadIdx.x] = q;
    __syncthreads();
    for (int off = 128; off > 0; off >>= 1) {
        if (threadIdx.x < off) { ls[threadIdx.x] += ls[threadIdx.x + off];
                                 lq[threadIdx.x] += lq[threadIdx.x + off]; }
        __syncthreads();
    }
    if (threadIdx.x == 0) { atomAddF(&lnS[0], ls[0]); atomAddF(&lnS[1], lq[0]); }
}

// ---------------------------------------------------------------------------
// Kernel 4: whole-tensor LayerNorm + softplus + per-graph add-pool / counts
//   PyG LN: xc = h - mean(h); out = xc / (sqrt(mean(xc^2)) + EPS) * w + b
// ---------------------------------------------------------------------------
__global__ __launch_bounds__(256)
void ln_pool(const float* __restrict__ h, const int* __restrict__ batch,
             const float* __restrict__ ln_w, const float* __restrict__ ln_b,
             const float* __restrict__ lnS, float* __restrict__ pool,
             float* __restrict__ cnt, long total, float invT)
{
    float mu  = lnS[0] * invT;
    float ms  = lnS[1] * invT - mu * mu;        // mean(xc^2)
    float den = sqrtf(fmaxf(ms, 0.0f)) + EPSV;
    long stride = (long)gridDim.x * blockDim.x;
    for (long i = (long)blockIdx.x * blockDim.x + threadIdx.x; i < total; i += stride) {
        int f = (int)(i & 63);
        long node = i >> 6;
        float v  = (h[i] - mu) / den * ln_w[f] + ln_b[f];
        float sp = softplusf(v);
        int g = batch[node];
        atomAddF(&pool[(long)g * 64 + f], sp);
        if (f == 0) atomAddF(&cnt[g], 1.0f);
    }
}

// ---------------------------------------------------------------------------
// Kernel 5: head: [mean_pool, add_pool] @ Wl + bl, then whole-tensor LN.
//   1024 outputs, 131k MACs — single workgroup, VALU.
// ---------------------------------------------------------------------------
__global__ __launch_bounds__(256)
void final_head(const float* __restrict__ pool, const float* __restrict__ cnt,
                const float* __restrict__ Wl, const float* __restrict__ bl,
                const float* __restrict__ w4, const float* __restrict__ b4,
                float* __restrict__ out)
{
    __shared__ float ov[Gnum * 2];
    __shared__ float rs[256], rq[256];
    int tid = threadIdx.x;
    for (int o = tid; o < Gnum * 2; o += 256) {
        int g = o >> 1, c = o & 1;
        float inv = 1.0f / fmaxf(cnt[g], 1.0f);
        float acc = bl[c];
        for (int j = 0; j < 64; ++j) {
            float ap = pool[g * 64 + j];
            acc += ap * inv * Wl[j * 2 + c];          // mean-pool half
            acc += ap *       Wl[(64 + j) * 2 + c];   // add-pool half
        }
        ov[o] = acc;
    }
    __syncthreads();
    float s = 0.0f, q = 0.0f;
    for (int o = tid; o < Gnum * 2; o += 256) { s += ov[o]; q += ov[o] * ov[o]; }
    rs[tid] = s; rq[tid] = q;
    __syncthreads();
    for (int off = 128; off > 0; off >>= 1) {
        if (tid < off) { rs[tid] += rs[tid + off]; rq[tid] += rq[tid + off]; }
        __syncthreads();
    }
    float mu  = rs[0] * (1.0f / (Gnum * 2));
    float ms  = rq[0] * (1.0f / (Gnum * 2)) - mu * mu;
    float den = sqrtf(fmaxf(ms, 0.0f)) + EPSV;
    for (int o = tid; o < Gnum * 2; o += 256) {
        int c = o & 1;
        out[o] = (ov[o] - mu) / den * w4[c] + b4[c];
    }
}

// ---------------------------------------------------------------------------
extern "C" void kernel_launch(void* const* d_in, const int* in_sizes, int n_in,
                              void* d_out, int out_size, void* d_ws, size_t ws_size,
                              hipStream_t stream)
{
    const float* x     = (const float*)d_in[0];
    const int*   ei    = (const int*)  d_in[1];
    const float* ea    = (const float*)d_in[2];
    const int*   batch = (const int*)  d_in[3];
    const float* Wf    = (const float*)d_in[4];
    const float* bfp   = (const float*)d_in[5];
    const float* Ws    = (const float*)d_in[6];
    const float* bsp   = (const float*)d_in[7];
    const float* bnw   = (const float*)d_in[8];
    const float* bnb   = (const float*)d_in[9];
    const float* lnw   = (const float*)d_in[10];
    const float* lnb   = (const float*)d_in[11];
    const float* Wl    = (const float*)d_in[12];
    const float* bl    = (const float*)d_in[13];
    const float* w4    = (const float*)d_in[14];
    const float* b4    = (const float*)d_in[15];
    float* out = (float*)d_out;

    const int N = in_sizes[0] / Fdim;
    const int E = in_sizes[1] / 2;

    const int* srcIdx = ei;          // edge_index[0]
    const int* dstIdx = ei + E;      // edge_index[1]

    // workspace layout (floats)
    float* ws     = (float*)d_ws;
    float* agg    = ws;                              // N*64
    float* hbuf   = agg    + (size_t)N * Fdim;       // N*64
    float* colsum = hbuf   + (size_t)N * Fdim;       // 64
    float* colsq  = colsum + Fdim;                   // 64
    float* lnS    = colsq  + Fdim;                   // 2 (+pad)
    float* pool   = lnS    + 64;                     // G*64
    float* cnt    = pool   + (size_t)Gnum * Fdim;    // G
    long zeroN = (long)((cnt + Gnum) - ws);

    zero_f32<<<2048, 256, 0, stream>>>(ws, zeroN);

    int eb = (E + 127) / 128;        // 8 waves/WG * 16 edges/wave
    edge_msg_wmma<<<eb, 256, 0, stream>>>(x, srcIdx, dstIdx, ea,
                                          Wf, bfp, Ws, bsp, agg, E);

    col_stats<<<256, 256, 0, stream>>>(agg, colsum, colsq, N);

    long total = (long)N * Fdim;
    bn_residual<<<2048, 256, 0, stream>>>(agg, x, colsum, colsq, bnw, bnb,
                                          hbuf, lnS, total, 1.0f / (float)N);

    ln_pool<<<2048, 256, 0, stream>>>(hbuf, batch, lnw, lnb, lnS,
                                      pool, cnt, total, 1.0f / (float)total);

    final_head<<<1, 256, 0, stream>>>(pool, cnt, Wl, bl, w4, b4, out);
}
